// MoSARA_10642928959975
// MI455X (gfx1250) — matrix operand order repaired and checked by reading, not compiled, not run on hardware
//
#include <hip/hip_runtime.h>
#include <hip/hip_bf16.h>
#include <math.h>

// ---------------------------------------------------------------------------
// MoSARA fused low-rank MoE adapter for gfx1250 (MI455X), wave32 + WMMA bf16.
//   out = gsum .* (x@W) + ((x@U_k) .* (g@lambda)) @ V_k^T, scaled by (1+v)
//   g   = softmax( (x @ (U_k@W_g1)) * W_g2 ),  gsum = sum_e g_e
// Big GEMMs run on V_WMMA_F32_16X16X32_BF16 (fp32 accumulate), 32x64 wave
// tiles, bf16 operands staged in workspace, 32-bit offsets + launch_bounds
// to keep the K-loop spill-free.
// ---------------------------------------------------------------------------

typedef __attribute__((ext_vector_type(16))) __bf16          bf16x16;
typedef __attribute__((ext_vector_type(8)))  float           f32x8;
typedef __attribute__((ext_vector_type(4)))  float           f32x4;
typedef __attribute__((ext_vector_type(8)))  unsigned short  u16x8;

__device__ __forceinline__ unsigned short f2bf(float f) {
  union { float f; unsigned u; } c; c.f = f;
  unsigned u = c.u;
  u += 0x7fffu + ((u >> 16) & 1u);          // round-to-nearest-even
  return (unsigned short)(u >> 16);
}
__device__ __forceinline__ unsigned pack2bf(float a, float b) {
  return (unsigned)f2bf(a) | ((unsigned)f2bf(b) << 16);
}

union Frag {                                 // 8 VGPRs = 16 bf16 per lane
  bf16x16 v;
  u16x8   h[2];
  unsigned u[8];
  unsigned short s[16];
};

// A fragment (16x32 bf16) from fp32 row-major source (fallback path).
__device__ __forceinline__ void load_a_f32(Frag& f, const float* base, unsigned off) {
  const f32x4* p0 = (const f32x4*)(base + off);
  const f32x4* p1 = (const f32x4*)(base + off + 16);
  f32x4 c0 = p0[0], c1 = p0[1];
  f32x4 d0 = p1[0], d1 = p1[1];
  f.u[0] = pack2bf(c0[0], c0[1]); f.u[1] = pack2bf(c0[2], c0[3]);
  f.u[2] = pack2bf(c1[0], c1[1]); f.u[3] = pack2bf(c1[2], c1[3]);
  f.u[4] = pack2bf(d0[0], d0[1]); f.u[5] = pack2bf(d0[2], d0[3]);
  f.u[6] = pack2bf(d1[0], d1[1]); f.u[7] = pack2bf(d1[2], d1[3]);
}

// A fragment from bf16 row-major source (two contiguous 16B loads, no cvt).
// off already includes row*ld + 8*hh; second chunk at +16 elements.
__device__ __forceinline__ void load_a_bf16(Frag& f, const unsigned short* base, unsigned off) {
  f.h[0] = *(const u16x8*)(base + off);
  f.h[1] = *(const u16x8*)(base + off + 16);
}

template <bool ABF16>
__device__ __forceinline__ void load_a(Frag& f, const void* base, unsigned off) {
  if constexpr (ABF16) load_a_bf16(f, (const unsigned short*)base, off);
  else                 load_a_f32(f, (const float*)base, off);
}

// B fragment (32x16 bf16): off includes row*ld + 8*hh; second chunk at +16 elements.
__device__ __forceinline__ void load_b(Frag& f, const unsigned short* base, unsigned off) {
  f.h[0] = *(const u16x8*)(base + off);
  f.h[1] = *(const u16x8*)(base + off + 16);
}

__device__ __forceinline__ f32x8 wmma_bf16(const Frag& a, const Frag& b, f32x8 c) {
  return __builtin_amdgcn_wmma_f32_16x16x32_bf16(false, a.v, false, b.v,
                                                 (short)0, c, false, false);
}

// ---------------------------------------------------------------------------
// u1[d] = sum_j U_k[d,j] * W_g1[j]
__global__ void k_u1(const float* __restrict__ U, const float* __restrict__ Wg1,
                     float* __restrict__ u1, int D, int k) {
  int d = blockIdx.x * blockDim.x + threadIdx.x;
  if (d >= D) return;
  const float* row = U + (size_t)d * k;
  float acc = 0.f;
  for (int j = 0; j < k; ++j) acc += row[j] * Wg1[j];
  u1[d] = acc;
}

// Routing: one wave per row. s = x[row].u1 ; g = softmax(s*Wg2); gsum.
__global__ void k_routing(const float* __restrict__ x, const float* __restrict__ u1,
                          const float* __restrict__ Wg2,
                          float* __restrict__ g, float* __restrict__ gsum,
                          int D, int E) {
  int lane = threadIdx.x & 31;
  int wv   = threadIdx.x >> 5;
  int row  = blockIdx.x * (blockDim.x >> 5) + wv;
  const float* xr = x + (size_t)row * D;
  float s = 0.f;
  for (int d = lane; d < D; d += 32) s += xr[d] * u1[d];
#pragma unroll
  for (int m = 16; m >= 1; m >>= 1) s += __shfl_xor(s, m, 32);
  if (lane == 0) {
    float l[8];
    float mx = -INFINITY;
    for (int e = 0; e < E; ++e) { l[e] = s * Wg2[e]; mx = fmaxf(mx, l[e]); }
    float S = 0.f;
    for (int e = 0; e < E; ++e) { l[e] = __expf(l[e] - mx); S += l[e]; }
    float inv = 1.f / S, gs = 0.f;
    for (int e = 0; e < E; ++e) { float ge = l[e] * inv; g[(size_t)row * E + e] = ge; gs += ge; }
    gsum[row] = gs;
  }
}

// x (fp32) -> xbf (bf16), 8 elements per thread.
__global__ void k_cvt_x(const float* __restrict__ x, unsigned short* __restrict__ xbf,
                        size_t total8) {
  size_t t = (size_t)blockIdx.x * blockDim.x + threadIdx.x;
  if (t >= total8) return;
  const f32x4* p = (const f32x4*)(x + t * 8);
  f32x4 a = p[0], b = p[1];
  union { u16x8 v; unsigned u[4]; } o;
  o.u[0] = pack2bf(a[0], a[1]); o.u[1] = pack2bf(a[2], a[3]);
  o.u[2] = pack2bf(b[0], b[1]); o.u[3] = pack2bf(b[2], b[3]);
  *((u16x8*)xbf + t) = o.v;
}

// LDS-tiled transpose + fp32->bf16: dst[c*dstLd + colOff + r] = src[r*C + c]
__global__ void k_transpose_cvt(const float* __restrict__ src, int R, int C,
                                unsigned short* __restrict__ dst, int dstLd,
                                int dstRows, int colOff) {
  __shared__ float tile[32][33];
  int c0 = blockIdx.x * 32, r0 = blockIdx.y * 32;
  int tx = threadIdx.x, ty = threadIdx.y;
#pragma unroll
  for (int i = 0; i < 4; ++i) {
    int r = r0 + ty + 8 * i, c = c0 + tx;
    tile[ty + 8 * i][tx] = (r < R && c < C) ? src[(size_t)r * C + c] : 0.f;
  }
  __syncthreads();
#pragma unroll
  for (int i = 0; i < 4; ++i) {
    int dr = c0 + ty + 8 * i, dc = r0 + tx;
    if (dr < dstRows && dc < R)
      dst[(size_t)dr * dstLd + colOff + dc] = f2bf(tile[tx][ty + 8 * i]);
  }
}

// BT[d, D+j] = bf16(V_k[d, j]) with zero pad for j >= k
__global__ void k_cvt_pad_v(const float* __restrict__ V, unsigned short* __restrict__ BT,
                            int ldBT, int D, int k, int kp) {
  size_t t = (size_t)blockIdx.x * blockDim.x + threadIdx.x;
  if (t >= (size_t)D * kp) return;
  int d = (int)(t / kp), j = (int)(t % kp);
  BT[(size_t)d * ldBT + D + j] = (j < k) ? f2bf(V[(size_t)d * k + j]) : (unsigned short)0;
}

// ---------------------------------------------------------------------------
// GEMM1: y[n,j] = (x @ U_k)[n,j] * coef[n,j], coef = g @ lambda. bf16 output.
// One wave computes a 32x64 tile; 8 WMMA per K=32 step.
template <bool ABF16>
__global__ __launch_bounds__(256, 1)
void k_gemm_xu_t(const void* __restrict__ A, const unsigned short* __restrict__ UT,
                 const float* __restrict__ g, const float* __restrict__ lam,
                 unsigned short* __restrict__ ybf,
                 int D, int kp, int k, int E, int tilesJ, int totalTiles) {
  int lane = threadIdx.x & 31;
  int tile = blockIdx.x * (blockDim.x >> 5) + (threadIdx.x >> 5);
  if (tile >= totalTiles) return;
  int tm = tile / tilesJ, tj = tile % tilesJ;
  int m0 = tm * 32, j0 = tj * 64;
  int n = lane & 15, hh = lane >> 4;

  // 32-bit element offsets (uniform bases stay in SGPRs)
  unsigned aOff0 = (unsigned)(m0 + n) * (unsigned)D + 8u * hh;
  unsigned aOff1 = aOff0 + 16u * (unsigned)D;
  unsigned bOff0 = (unsigned)(j0 + n) * (unsigned)D + 8u * hh;
  unsigned bStep = 16u * (unsigned)D;       // 16 rows of UT

  f32x8 acc[2][4] = {};
  Frag a[2], b[4];
  for (int kb = 0; kb < D; kb += 32) {
    __builtin_prefetch(UT + bOff0 + kb + 128, 0, 3);
    __builtin_prefetch(UT + bOff0 + 2 * bStep + kb + 128, 0, 3);
    load_a<ABF16>(a[0], A, aOff0 + kb);
    load_a<ABF16>(a[1], A, aOff1 + kb);
#pragma unroll
    for (int jb = 0; jb < 4; ++jb) load_b(b[jb], UT, bOff0 + jb * bStep + kb);
#pragma unroll
    for (int i = 0; i < 2; ++i)
#pragma unroll
      for (int jb = 0; jb < 4; ++jb)
        acc[i][jb] = wmma_bf16(a[i], b[jb], acc[i][jb]);
  }

  // epilogue: multiply by coef[n,j] = sum_e g[n,e]*lambda[e,j], store bf16
  float lamv[4][8];
#pragma unroll
  for (int jb = 0; jb < 4; ++jb) {
    int j = j0 + 16 * jb + n;
    for (int e = 0; e < E; ++e)
      lamv[jb][e] = (j < k) ? lam[(size_t)e * k + j] : 0.f;
  }
#pragma unroll
  for (int r = 0; r < 8; ++r) {
    int row0 = m0 + r + 8 * hh;
    int row1 = row0 + 16;
    float c0[4] = {0.f, 0.f, 0.f, 0.f};
    float c1[4] = {0.f, 0.f, 0.f, 0.f};
    for (int e = 0; e < E; ++e) {
      float g0 = g[(size_t)row0 * E + e];
      float g1 = g[(size_t)row1 * E + e];
#pragma unroll
      for (int jb = 0; jb < 4; ++jb) { c0[jb] += g0 * lamv[jb][e]; c1[jb] += g1 * lamv[jb][e]; }
    }
#pragma unroll
    for (int jb = 0; jb < 4; ++jb) {
      int j = j0 + 16 * jb + n;
      ybf[(size_t)row0 * kp + j] = f2bf(acc[0][jb][r] * c0[jb]);
      ybf[(size_t)row1 * kp + j] = f2bf(acc[1][jb][r] * c1[jb]);
    }
  }
}

// ---------------------------------------------------------------------------
// GEMM2: out[n,d] = (gsum[n]*(x@W)[n,d] + (y@V^T)[n,d]) * (1+v[d])
// BT holds [ W^T | V_k ] bf16, K contiguous per row; shared accumulators
// across both phases, per-row gsum scale applied between phases. 32x64 tile.
template <bool ABF16>
__global__ __launch_bounds__(256, 1)
void k_gemm_out_t(const void* __restrict__ A, const unsigned short* __restrict__ ybf,
                  const unsigned short* __restrict__ BT,
                  const float* __restrict__ gsum, const float* __restrict__ v,
                  float* __restrict__ out,
                  int D, int kp, int ldBT, int tilesN, int totalTiles) {
  int lane = threadIdx.x & 31;
  int tile = blockIdx.x * (blockDim.x >> 5) + (threadIdx.x >> 5);
  if (tile >= totalTiles) return;
  int tm = tile / tilesN, tn = tile % tilesN;
  int m0 = tm * 32, d0 = tn * 64;
  int n = lane & 15, hh = lane >> 4;

  unsigned aOff0 = (unsigned)(m0 + n) * (unsigned)D + 8u * hh;
  unsigned aOff1 = aOff0 + 16u * (unsigned)D;
  unsigned bOff0 = (unsigned)(d0 + n) * (unsigned)ldBT + 8u * hh;
  unsigned bStep = 16u * (unsigned)ldBT;    // 16 rows of BT

  f32x8 acc[2][4] = {};
  Frag a[2], b[4];

  // phase 1: x @ W  (K = D)
  for (int kb = 0; kb < D; kb += 32) {
    __builtin_prefetch(BT + bOff0 + kb + 128, 0, 3);
    __builtin_prefetch(BT + bOff0 + 2 * bStep + kb + 128, 0, 3);
    load_a<ABF16>(a[0], A, aOff0 + kb);
    load_a<ABF16>(a[1], A, aOff1 + kb);
#pragma unroll
    for (int jb = 0; jb < 4; ++jb) load_b(b[jb], BT, bOff0 + jb * bStep + kb);
#pragma unroll
    for (int i = 0; i < 2; ++i)
#pragma unroll
      for (int jb = 0; jb < 4; ++jb)
        acc[i][jb] = wmma_bf16(a[i], b[jb], acc[i][jb]);
  }

  // per-row gate-sum scale of the dense part
#pragma unroll
  for (int r = 0; r < 8; ++r) {
    float gs0 = gsum[m0 + r + 8 * hh];
    float gs1 = gsum[m0 + 16 + r + 8 * hh];
#pragma unroll
    for (int jb = 0; jb < 4; ++jb) { acc[0][jb][r] *= gs0; acc[1][jb][r] *= gs1; }
  }

  // phase 2: y @ V^T  (K = kp), same accumulators
  unsigned yOff0 = (unsigned)(m0 + n) * (unsigned)kp + 8u * hh;
  unsigned yOff1 = yOff0 + 16u * (unsigned)kp;
  for (int kb = 0; kb < kp; kb += 32) {
    __builtin_prefetch(ybf + yOff0 + kb + 32, 0, 3);
    load_a_bf16(a[0], ybf, yOff0 + kb);
    load_a_bf16(a[1], ybf, yOff1 + kb);
#pragma unroll
    for (int jb = 0; jb < 4; ++jb) load_b(b[jb], BT, bOff0 + jb * bStep + (unsigned)D + kb);
#pragma unroll
    for (int i = 0; i < 2; ++i)
#pragma unroll
      for (int jb = 0; jb < 4; ++jb)
        acc[i][jb] = wmma_bf16(a[i], b[jb], acc[i][jb]);
  }

  float vs[4];
#pragma unroll
  for (int jb = 0; jb < 4; ++jb) vs[jb] = 1.f + v[d0 + 16 * jb + n];
#pragma unroll
  for (int r = 0; r < 8; ++r) {
    int row0 = m0 + r + 8 * hh;
    int row1 = row0 + 16;
#pragma unroll
    for (int jb = 0; jb < 4; ++jb) {
      int d = d0 + 16 * jb + n;
      out[(size_t)row0 * D + d] = acc[0][jb][r] * vs[jb];
      out[(size_t)row1 * D + d] = acc[1][jb][r] * vs[jb];
    }
  }
}

// ---------------------------------------------------------------------------
extern "C" void kernel_launch(void* const* d_in, const int* in_sizes, int n_in,
                              void* d_out, int out_size, void* d_ws, size_t ws_size,
                              hipStream_t stream) {
  const float* x   = (const float*)d_in[0];
  const float* W   = (const float*)d_in[1];
  const float* U   = (const float*)d_in[2];
  const float* V   = (const float*)d_in[3];
  const float* lam = (const float*)d_in[4];
  const float* v   = (const float*)d_in[5];
  const float* Wg1 = (const float*)d_in[6];
  const float* Wg2 = (const float*)d_in[7];
  float* out = (float*)d_out;

  const int D  = in_sizes[5];
  const int N  = in_sizes[0] / D;
  const int k  = in_sizes[2] / D;
  const int E  = in_sizes[7];
  const int kp = ((k + 63) / 64) * 64;   // pad rank to 64 for 32x64 tiles
  const int ldBT = D + kp;
  if (D <= 0 || N <= 0 || k <= 0) return;

  // workspace carve (256B aligned)
  char* p = (char*)d_ws;
  auto carve = [&](size_t bytes) -> void* {
    void* r = (void*)p;
    p += (bytes + 255) & ~(size_t)255;
    return r;
  };
  float*          u1   = (float*)carve((size_t)D * 4);
  float*          g    = (float*)carve((size_t)N * E * 4);
  float*          gsum = (float*)carve((size_t)N * 4);
  unsigned short* BT   = (unsigned short*)carve((size_t)D * ldBT * 2);
  unsigned short* UT   = (unsigned short*)carve((size_t)kp * D * 2);
  unsigned short* ybf  = (unsigned short*)carve((size_t)N * kp * 2);
  size_t usedBase = (size_t)(p - (char*)d_ws);
  size_t xbfBytes = (size_t)N * D * 2;
  bool useXbf = (usedBase + xbfBytes + 256) <= ws_size;
  unsigned short* xbf = useXbf ? (unsigned short*)carve(xbfBytes) : nullptr;

  // 1) u1 = U_k @ W_g1
  k_u1<<<(D + 255) / 256, 256, 0, stream>>>(U, Wg1, u1, D, k);

  // 2) routing (one wave per row, 8 rows per 256-thread block)
  k_routing<<<N / 8, 256, 0, stream>>>(x, u1, Wg2, g, gsum, D, E);

  // 3) layout: BT[:, 0:D] = W^T ; BT[:, D:D+kp] = V_k (padded) ; UT = U_k^T (padded)
  {
    dim3 blk(32, 8);
    dim3 gw((D + 31) / 32, (D + 31) / 32);
    k_transpose_cvt<<<gw, blk, 0, stream>>>(W, D, D, BT, ldBT, D, 0);
    dim3 gu((kp + 31) / 32, (D + 31) / 32);
    k_transpose_cvt<<<gu, blk, 0, stream>>>(U, D, k, UT, D, kp, 0);
    size_t tot = (size_t)D * kp;
    k_cvt_pad_v<<<(unsigned)((tot + 255) / 256), 256, 0, stream>>>(V, BT, ldBT, D, k, kp);
  }

  // 3b) x -> bf16 (fast-path A operand)
  if (useXbf) {
    size_t total8 = (size_t)N * D / 8;
    k_cvt_x<<<(unsigned)((total8 + 255) / 256), 256, 0, stream>>>(x, xbf, total8);
  }

  // 4) GEMM1: y = (x@U_k) .* (g@lambda), bf16
  {
    int tilesJ = kp / 64;
    int tiles  = (N / 32) * tilesJ;
    int blocks = (tiles + 7) / 8;
    if (useXbf)
      k_gemm_xu_t<true><<<blocks, 256, 0, stream>>>((const void*)xbf, UT, g, lam, ybf,
                                                    D, kp, k, E, tilesJ, tiles);
    else
      k_gemm_xu_t<false><<<blocks, 256, 0, stream>>>((const void*)x, UT, g, lam, ybf,
                                                     D, kp, k, E, tilesJ, tiles);
  }

  // 5) GEMM2: out = (gsum.*(x@W) + y@V^T) .* (1+v)
  {
    int tilesN = D / 64;
    int tiles  = (N / 32) * tilesN;
    int blocks = (tiles + 7) / 8;
    if (useXbf)
      k_gemm_out_t<true><<<blocks, 256, 0, stream>>>((const void*)xbf, ybf, BT, gsum, v, out,
                                                     D, kp, ldBT, tilesN, tiles);
    else
      k_gemm_out_t<false><<<blocks, 256, 0, stream>>>((const void*)x, ybf, BT, gsum, v, out,
                                                      D, kp, ldBT, tilesN, tiles);
  }
}
